// memory_65292092834159
// MI455X (gfx1250) — matrix operand order repaired and checked by reading, not compiled, not run on hardware
//
#include <hip/hip_runtime.h>
#include <hip/hip_bf16.h>
#include <math.h>

#define DEV __device__ __forceinline__

typedef __attribute__((ext_vector_type(16))) _Float16 v16h;
typedef __attribute__((ext_vector_type(8)))  _Float16 v8h;
typedef __attribute__((ext_vector_type(8)))  float    v8f;
typedef __attribute__((ext_vector_type(4)))  float    v4f;

static constexpr int B_   = 4096;
static constexpr int DEG  = 50;
static constexpr int FEAT = 256;
static constexpr int D_   = 128;
static constexpr int ROWS = B_ * DEG;   // 204800 (multiple of 16)

// ---------------------------------------------------------------------------
// WMMA fragment helpers (CDNA5 wave32, V_WMMA_F32_16X16X32_F16 layouts)
// A 16x32 f16: lane<16 holds M=lane, elements e: K = e<8 ? kb+e     : kb+8+e
//              lane>=16 holds M=lane-16,         K = e<8 ? kb+8+e   : kb+16+e
// B 32x16 f16: lane n (N=n&15), element e: same K as A (e<8 ? kb+8h+e : kb+16+8h+(e-8))
// C/D 16x16 f32: element r: M = r + (lane>=16 ? 8 : 0), N = lane&15
// ---------------------------------------------------------------------------

DEV v8f wmma_f16(v16h a, v16h b, v8f c) {
  return __builtin_amdgcn_wmma_f32_16x16x32_f16(
      /*neg_a=*/false, a, /*neg_b=*/false, b,
      /*c_mod=*/(short)0, c, /*reuse_a=*/false, /*reuse_b=*/false);
}

// A-fragment from an f32 row (row pointer already selected per lane's M)
DEV v16h load_a_f32(const float* __restrict__ rowp, int kbase, bool hi) {
  const int k0 = kbase + (hi ? 8 : 0);
  const v4f f0 = *(const v4f*)(rowp + k0);
  const v4f f1 = *(const v4f*)(rowp + k0 + 4);
  const v4f f2 = *(const v4f*)(rowp + k0 + 16);
  const v4f f3 = *(const v4f*)(rowp + k0 + 20);
  v16h a;
#pragma unroll
  for (int i = 0; i < 4; ++i) {
    a[i]      = (_Float16)f0[i];
    a[4 + i]  = (_Float16)f1[i];
    a[8 + i]  = (_Float16)f2[i];
    a[12 + i] = (_Float16)f3[i];
  }
  return a;
}

// A-fragment from an f16 row (global)
DEV v16h load_a_f16(const _Float16* __restrict__ rowp, int kbase, bool hi) {
  const int k0 = kbase + (hi ? 8 : 0);
  const v8h lo = *(const v8h*)(rowp + k0);
  const v8h hh = *(const v8h*)(rowp + k0 + 16);
  v16h a;
#pragma unroll
  for (int i = 0; i < 8; ++i) { a[i] = lo[i]; a[8 + i] = hh[i]; }
  return a;
}

// B-fragment from packed transposed weights Bt[N][K] (f16, row-major in N)
DEV v16h load_b16(const _Float16* __restrict__ Bt, int Krows, int nGlob,
                  int kbase, bool hi) {
  const _Float16* p = Bt + (size_t)nGlob * Krows + kbase + (hi ? 8 : 0);
  const v8h lo = *(const v8h*)(p);
  const v8h hh = *(const v8h*)(p + 16);
  v16h b;
#pragma unroll
  for (int i = 0; i < 8; ++i) { b[i] = lo[i]; b[8 + i] = hh[i]; }
  return b;
}

// ---------------------------------------------------------------------------
// Weight pack: W[K][N] f32  ->  Wt[N][K] f16
// ---------------------------------------------------------------------------
__global__ void pack_wT(const float* __restrict__ W, _Float16* __restrict__ Wt,
                        int K, int N) {
  int idx = blockIdx.x * 256 + threadIdx.x;
  if (idx >= N * K) return;
  int n = idx / K, k = idx - n * K;
  Wt[idx] = (_Float16)W[(size_t)k * N + n];
}

// ---------------------------------------------------------------------------
// GEMM (gathered f32 rows): out[r][:] = ufeat[idx[r]] @ W + bias, f16 out
// K=256, N=128. One wave -> 16 rows x 128 cols.  256 thr = 8 waves = 8 tiles.
// ---------------------------------------------------------------------------
__global__ __launch_bounds__(256)
void gemm_gather_f32(const float* __restrict__ ufeat,
                     const int* __restrict__ idx,
                     const _Float16* __restrict__ BT,   // [128 x 256]
                     const float* __restrict__ bias,    // [128]
                     _Float16* __restrict__ outH,       // [rows x 128]
                     int nTiles) {
  const int tile = blockIdx.x * 8 + (threadIdx.x >> 5);
  if (tile >= nTiles) return;
  const int  lane = threadIdx.x & 31;
  const bool hi   = lane >= 16;
  const int  nloc = lane & 15;
  const int  rowBase = tile * 16;
  const float* rowp = ufeat + (size_t)idx[rowBase + nloc] * FEAT;
  // get the random 1KB L2-resident gather row in flight early
  __builtin_prefetch(rowp, 0, 3);
  __builtin_prefetch(rowp + 128, 0, 3);

  v8f acc[8];
#pragma unroll
  for (int nt = 0; nt < 8; ++nt) acc[nt] = v8f{};

#pragma unroll
  for (int ks = 0; ks < 8; ++ks) {
    v16h a = load_a_f32(rowp, ks * 32, hi);
#pragma unroll
    for (int nt = 0; nt < 8; ++nt) {
      v16h b = load_b16(BT, 256, nt * 16 + nloc, ks * 32, hi);
      acc[nt] = wmma_f16(a, b, acc[nt]);
    }
  }
#pragma unroll
  for (int nt = 0; nt < 8; ++nt) {
    const int col = nt * 16 + nloc;
    const float bv = bias[col];
#pragma unroll
    for (int r = 0; r < 8; ++r) {
      const int orow = rowBase + r + (hi ? 8 : 0);
      outH[(size_t)orow * D_ + col] = (_Float16)(acc[nt][r] + bv);
    }
  }
}

// ---------------------------------------------------------------------------
// att_j: per row, logits[k] = sum_d nf[b][d]*ff[r][d]*K[d][k]; softmax over 4
// One wave per row.
// ---------------------------------------------------------------------------
__global__ __launch_bounds__(256)
void att_j_kernel(const _Float16* __restrict__ ffH,
                  const _Float16* __restrict__ nfH,
                  const float* __restrict__ Kmat,      // [128 x 4]
                  float* __restrict__ att) {           // [ROWS x 4]
  const int row = blockIdx.x * 8 + (threadIdx.x >> 5);
  if (row >= ROWS) return;
  const int lane = threadIdx.x & 31;
  const int b = row / DEG;
  float s0 = 0.f, s1 = 0.f, s2 = 0.f, s3 = 0.f;
  for (int d = lane; d < D_; d += 32) {
    const float c = (float)ffH[(size_t)row * D_ + d] * (float)nfH[(size_t)b * D_ + d];
    const v4f kv = *(const v4f*)(Kmat + d * 4);
    s0 += c * kv[0]; s1 += c * kv[1]; s2 += c * kv[2]; s3 += c * kv[3];
  }
#pragma unroll
  for (int off = 16; off; off >>= 1) {
    s0 += __shfl_xor(s0, off, 32);
    s1 += __shfl_xor(s1, off, 32);
    s2 += __shfl_xor(s2, off, 32);
    s3 += __shfl_xor(s3, off, 32);
  }
  if (lane == 0) {
    float m = fmaxf(fmaxf(s0, s1), fmaxf(s2, s3));
    float e0 = expf(s0 - m), e1 = expf(s1 - m), e2 = expf(s2 - m), e3 = expf(s3 - m);
    float inv = 1.f / (e0 + e1 + e2 + e3);
    float* p = att + (size_t)row * 4;
    p[0] = e0 * inv; p[1] = e1 * inv; p[2] = e2 * inv; p[3] = e3 * inv;
  }
}

// ---------------------------------------------------------------------------
// fil: fuse V = relu(ff @ W2 + b2) with fil = sum_k att[k]*V[:,k*128:+128].
// One block (8 waves) per 16-row tile; wave w owns fil cols [w*16, w*16+16),
// computing the 4 V tiles (k=0..3) that fold into those columns.
// The shared 16x128 A tile is staged once in LDS (padded stride: 16 rows hit
// distinct banks) and the 4 A fragments are loaded once and reused over k.
// ---------------------------------------------------------------------------
__global__ __launch_bounds__(256)
void fil_kernel(const _Float16* __restrict__ ffH,
                const _Float16* __restrict__ W2T,     // [512 x 128]
                const float* __restrict__ b2,         // [512]
                const float* __restrict__ att,        // [ROWS x 4]
                _Float16* __restrict__ filH) {        // [ROWS x 128]
  constexpr int LSTR = 136;                 // 272B row stride: 68 dwords -> bank offset 4r
  __shared__ _Float16 ff_s[16 * LSTR];
  __shared__ float att_s[16 * 4];
  const int rowBase = blockIdx.x * 16;
  {
    const int t = threadIdx.x;              // 256 threads x 16B = 4KB tile
    const int row = t >> 4, chunk = (t & 15) * 8;
    v8h v = *(const v8h*)(ffH + (size_t)(rowBase + row) * D_ + chunk);
    *(v8h*)(ff_s + row * LSTR + chunk) = v;
    if (t < 64) att_s[t] = att[(size_t)rowBase * 4 + t];
  }
  __syncthreads();

  const int  w    = threadIdx.x >> 5;
  const int  lane = threadIdx.x & 31;
  const bool hi   = lane >= 16;
  const int  nloc = lane & 15;
  const int  dcol = w * 16 + nloc;
  const _Float16* rowp = ff_s + nloc * LSTR;

  // A fragments from LDS, loaded once, reused for all 4 k-tiles
  v16h afrag[4];
#pragma unroll
  for (int ks = 0; ks < 4; ++ks) {
    const int k0 = ks * 32 + (hi ? 8 : 0);
    const v8h lo = *(const v8h*)(rowp + k0);
    const v8h hh = *(const v8h*)(rowp + k0 + 16);
#pragma unroll
    for (int i = 0; i < 8; ++i) { afrag[ks][i] = lo[i]; afrag[ks][8 + i] = hh[i]; }
  }

  v8f fac = v8f{};
#pragma unroll
  for (int k = 0; k < 4; ++k) {
    const int nGlob = k * 128 + dcol;            // V column
    v8f c = v8f{};
#pragma unroll
    for (int ks = 0; ks < 4; ++ks) {
      v16h b = load_b16(W2T, 128, nGlob, ks * 32, hi);
      c = wmma_f16(afrag[ks], b, c);
    }
    const float bv = b2[nGlob];
#pragma unroll
    for (int r = 0; r < 8; ++r) {
      float v = c[r] + bv;
      v = v > 0.f ? v : 0.f;
      const int m = r + (hi ? 8 : 0);
      fac[r] += att_s[m * 4 + k] * v;
    }
  }
#pragma unroll
  for (int r = 0; r < 8; ++r)
    filH[(size_t)(rowBase + r + (hi ? 8 : 0)) * D_ + dcol] = (_Float16)fac[r];
}

// ---------------------------------------------------------------------------
// x1 = relu(concat([fil, nf_broadcast]) @ Wa1 + ba1), K=256
// K-steps 0..3 read fil rows, 4..7 read node_feat[b] (b = row/DEG).
// ---------------------------------------------------------------------------
__global__ __launch_bounds__(256)
void gemm_cat_relu(const _Float16* __restrict__ filH,
                   const _Float16* __restrict__ nfH,
                   const _Float16* __restrict__ BT,    // [128 x 256]
                   const float* __restrict__ bias,
                   _Float16* __restrict__ outH, int nTiles) {
  const int tile = blockIdx.x * 8 + (threadIdx.x >> 5);
  if (tile >= nTiles) return;
  const int  lane = threadIdx.x & 31;
  const bool hi   = lane >= 16;
  const int  nloc = lane & 15;
  const int  rowBase = tile * 16;
  const int  myRow = rowBase + nloc;
  const _Float16* r1 = filH + (size_t)myRow * D_;
  const _Float16* r2 = nfH + (size_t)(myRow / DEG) * D_;

  v8f acc[8];
#pragma unroll
  for (int nt = 0; nt < 8; ++nt) acc[nt] = v8f{};

#pragma unroll
  for (int ks = 0; ks < 8; ++ks) {
    const _Float16* rp = (ks < 4) ? r1 : r2;
    v16h a = load_a_f16(rp, (ks & 3) * 32, hi);
#pragma unroll
    for (int nt = 0; nt < 8; ++nt) {
      v16h b = load_b16(BT, 256, nt * 16 + nloc, ks * 32, hi);
      acc[nt] = wmma_f16(a, b, acc[nt]);
    }
  }
#pragma unroll
  for (int nt = 0; nt < 8; ++nt) {
    const int col = nt * 16 + nloc;
    const float bv = bias[col];
#pragma unroll
    for (int r = 0; r < 8; ++r) {
      float v = acc[nt][r] + bv;
      v = v > 0.f ? v : 0.f;
      outH[(size_t)(rowBase + r + (hi ? 8 : 0)) * D_ + col] = (_Float16)v;
    }
  }
}

// ---------------------------------------------------------------------------
// x2 = relu(x1 @ Wa2 + ba2), K=128
// ---------------------------------------------------------------------------
__global__ __launch_bounds__(256)
void gemm_relu_128(const _Float16* __restrict__ A,
                   const _Float16* __restrict__ BT,    // [128 x 128]
                   const float* __restrict__ bias,
                   _Float16* __restrict__ outH, int nTiles) {
  const int tile = blockIdx.x * 8 + (threadIdx.x >> 5);
  if (tile >= nTiles) return;
  const int  lane = threadIdx.x & 31;
  const bool hi   = lane >= 16;
  const int  nloc = lane & 15;
  const int  rowBase = tile * 16;
  const _Float16* rowp = A + (size_t)(rowBase + nloc) * D_;

  v8f acc[8];
#pragma unroll
  for (int nt = 0; nt < 8; ++nt) acc[nt] = v8f{};

#pragma unroll
  for (int ks = 0; ks < 4; ++ks) {
    v16h a = load_a_f16(rowp, ks * 32, hi);
#pragma unroll
    for (int nt = 0; nt < 8; ++nt) {
      v16h b = load_b16(BT, 128, nt * 16 + nloc, ks * 32, hi);
      acc[nt] = wmma_f16(a, b, acc[nt]);
    }
  }
#pragma unroll
  for (int nt = 0; nt < 8; ++nt) {
    const int col = nt * 16 + nloc;
    const float bv = bias[col];
#pragma unroll
    for (int r = 0; r < 8; ++r) {
      float v = acc[nt][r] + bv;
      v = v > 0.f ? v : 0.f;
      outH[(size_t)(rowBase + r + (hi ? 8 : 0)) * D_ + col] = (_Float16)v;
    }
  }
}

// ---------------------------------------------------------------------------
// logits[r] = x2[r] . Wa3 + ba3   (one wave per row)
// ---------------------------------------------------------------------------
__global__ __launch_bounds__(256)
void logits_kernel(const _Float16* __restrict__ x2H,
                   const float* __restrict__ Wa3,      // [128]
                   const float* __restrict__ ba3,      // [1]
                   float* __restrict__ logits) {       // [ROWS]
  const int row = blockIdx.x * 8 + (threadIdx.x >> 5);
  if (row >= ROWS) return;
  const int lane = threadIdx.x & 31;
  float s = 0.f;
  for (int d = lane; d < D_; d += 32)
    s += (float)x2H[(size_t)row * D_ + d] * Wa3[d];
#pragma unroll
  for (int off = 16; off; off >>= 1) s += __shfl_xor(s, off, 32);
  if (lane == 0) logits[row] = s + ba3[0];
}

// ---------------------------------------------------------------------------
// out[b] = sum_f softmax_f(logits[b])[f] * fil[b,f]   (one block per b)
// ---------------------------------------------------------------------------
__global__ __launch_bounds__(128)
void out_kernel(const _Float16* __restrict__ filH,
                const float* __restrict__ logits,
                float* __restrict__ out) {             // [B x 128]
  __shared__ float p[DEG];
  const int b = blockIdx.x;
  const int t = threadIdx.x;
  if (t < DEG) p[t] = logits[(size_t)b * DEG + t];
  __syncthreads();
  if (t == 0) {
    float m = -3.4e38f;
    for (int f = 0; f < DEG; ++f) m = fmaxf(m, p[f]);
    float s = 0.f;
    for (int f = 0; f < DEG; ++f) { p[f] = expf(p[f] - m); s += p[f]; }
    const float inv = 1.f / s;
    for (int f = 0; f < DEG; ++f) p[f] *= inv;
  }
  __syncthreads();
  float acc = 0.f;
  for (int f = 0; f < DEG; ++f)
    acc += p[f] * (float)filH[(size_t)(b * DEG + f) * D_ + t];
  out[(size_t)b * D_ + t] = acc;
}

// ---------------------------------------------------------------------------
extern "C" void kernel_launch(void* const* d_in, const int* in_sizes, int n_in,
                              void* d_out, int out_size, void* d_ws, size_t ws_size,
                              hipStream_t stream) {
  const int*   nodes_u = (const int*)  d_in[0];
  const int*   friends = (const int*)  d_in[2];
  const float* ufeat   = (const float*)d_in[4];
  const float* Wu      = (const float*)d_in[5];
  const float* bu      = (const float*)d_in[6];
  const float* Wuf     = (const float*)d_in[7];
  const float* buf     = (const float*)d_in[8];
  const float* W2      = (const float*)d_in[9];
  const float* b2      = (const float*)d_in[10];
  const float* Kmat    = (const float*)d_in[11];
  const float* Wa1     = (const float*)d_in[12];
  const float* ba1     = (const float*)d_in[13];
  const float* Wa2     = (const float*)d_in[14];
  const float* ba2     = (const float*)d_in[15];
  const float* Wa3     = (const float*)d_in[16];
  const float* ba3     = (const float*)d_in[17];
  float* out = (float*)d_out;

  char* ws = (char*)d_ws;
  size_t off = 0;
  auto take = [&](size_t bytes) -> char* {
    char* p = ws + off;
    off += (bytes + 255) & ~(size_t)255;
    return p;
  };
  _Float16* WuT  = (_Float16*)take((size_t)D_ * FEAT * 2);        // [128 x 256]
  _Float16* WufT = (_Float16*)take((size_t)D_ * FEAT * 2);        // [128 x 256]
  _Float16* W2T  = (_Float16*)take((size_t)(4 * D_) * D_ * 2);    // [512 x 128]
  _Float16* Wa1T = (_Float16*)take((size_t)D_ * (2 * D_) * 2);    // [128 x 256]
  _Float16* Wa2T = (_Float16*)take((size_t)D_ * D_ * 2);          // [128 x 128]
  _Float16* nfH  = (_Float16*)take((size_t)B_ * D_ * 2);
  _Float16* ffH  = (_Float16*)take((size_t)ROWS * D_ * 2);
  float*    att  = (float*)   take((size_t)ROWS * 4 * 4);
  _Float16* filH = (_Float16*)take((size_t)ROWS * D_ * 2);
  _Float16* x1H  = (_Float16*)take((size_t)ROWS * D_ * 2);
  _Float16* x2H  = (_Float16*)take((size_t)ROWS * D_ * 2);
  float*    lgt  = (float*)   take((size_t)ROWS * 4);
  (void)ws_size; (void)in_sizes; (void)n_in; (void)out_size;

  // 1. pack weights (f32 [K][N] -> f16 [N][K])
  pack_wT<<<(FEAT * D_ + 255) / 256, 256, 0, stream>>>(Wu,  WuT,  FEAT, D_);
  pack_wT<<<(FEAT * D_ + 255) / 256, 256, 0, stream>>>(Wuf, WufT, FEAT, D_);
  pack_wT<<<(D_ * 4 * D_ + 255) / 256, 256, 0, stream>>>(W2,  W2T,  D_, 4 * D_);
  pack_wT<<<(2 * D_ * D_ + 255) / 256, 256, 0, stream>>>(Wa1, Wa1T, 2 * D_, D_);
  pack_wT<<<(D_ * D_ + 255) / 256, 256, 0, stream>>>(Wa2, Wa2T, D_, D_);

  // 2. node_feat = ufeat[nodes_u] @ Wu + bu  (256 tiles)
  gemm_gather_f32<<<(B_ / 16 + 7) / 8, 256, 0, stream>>>(
      ufeat, nodes_u, WuT, bu, nfH, B_ / 16);

  // 3. ff = ufeat[friends] @ Wuf + buf  (12800 tiles)
  gemm_gather_f32<<<(ROWS / 16 + 7) / 8, 256, 0, stream>>>(
      ufeat, friends, WufT, buf, ffH, ROWS / 16);

  // 4. att_j = softmax((nf * ff) @ K)
  att_j_kernel<<<ROWS / 8, 256, 0, stream>>>(ffH, nfH, Kmat, att);

  // 5. fil = einsum(att_j, relu(ff @ W2 + b2).reshape(.,4,D))
  fil_kernel<<<ROWS / 16, 256, 0, stream>>>(ffH, W2T, b2, att, filH);

  // 6. x1 = relu([fil ; nf] @ Wa1 + ba1)
  gemm_cat_relu<<<(ROWS / 16 + 7) / 8, 256, 0, stream>>>(
      filH, nfH, Wa1T, ba1, x1H, ROWS / 16);

  // 7. x2 = relu(x1 @ Wa2 + ba2)
  gemm_relu_128<<<(ROWS / 16 + 7) / 8, 256, 0, stream>>>(
      x1H, Wa2T, ba2, x2H, ROWS / 16);

  // 8. logits = x2 @ Wa3 + ba3
  logits_kernel<<<ROWS / 8, 256, 0, stream>>>(x2H, Wa3, ba3, lgt);

  // 9. out = sum_f softmax_f(logits) * fil
  out_kernel<<<B_, 128, 0, stream>>>(filH, lgt, out);
}
